// SPECTRELayer_27255862460984
// MI455X (gfx1250) — compile-verified, matrix operand order checked by reading, not run on hardware
//
#include <hip/hip_runtime.h>
#include <hip/hip_bf16.h>
#include <math.h>

// ---------------------------------------------------------------------------
// SPECTRE layer for MI455X (gfx1250), wave32 + WMMA bf16.
// B=8, L=2048, D=1024, H=16, DH=64, NF=1025.
// ---------------------------------------------------------------------------

#define B_ 8
#define L_ 2048
#define D_ 1024
#define H_ 16
#define DH_ 64
#define NF_ 1025
#define M_ (B_ * L_)        // 16384 rows for the big GEMMs

typedef __attribute__((ext_vector_type(16))) __bf16 v16bf;
typedef __attribute__((ext_vector_type(8)))  float  v8f;

// ------------------------------ small helpers ------------------------------

__device__ __forceinline__ float gelu_exact(float x) {
    return 0.5f * x * (1.0f + erff(x * 0.70710678118654752f));
}

// f32 -> bf16 cast (grid-stride)
__global__ void cast_f32_to_bf16(const float* __restrict__ in,
                                 __bf16* __restrict__ out, int n) {
    int i = blockIdx.x * blockDim.x + threadIdx.x;
    int stride = gridDim.x * blockDim.x;
    for (; i < n; i += stride) out[i] = (__bf16)in[i];
}

// xbar[b,k] = mean over l of x[b,l,k]   (8192 outputs)
__global__ void colmean_kernel(const float* __restrict__ x,
                               float* __restrict__ xbar) {
    int o = blockIdx.x * blockDim.x + threadIdx.x;
    if (o >= B_ * D_) return;
    int b = o >> 10, k = o & (D_ - 1);
    const float* p = x + (size_t)b * L_ * D_ + k;
    float s = 0.f;
    for (int l = 0; l < L_; ++l) s += p[(size_t)l * D_];
    xbar[o] = s * (1.0f / (float)L_);
}

// mean_q[b,n] = dot(xbar[b,:], Wq[n,:])   (mean commutes with linear map)
__global__ void meanq_kernel(const float* __restrict__ xbar,
                             const float* __restrict__ Wq,
                             float* __restrict__ meanq) {
    int o = blockIdx.x * blockDim.x + threadIdx.x;
    if (o >= B_ * D_) return;
    int b = o >> 10, n = o & (D_ - 1);
    const float* xb = xbar + b * D_;
    const float* w  = Wq + (size_t)n * D_;
    float s = 0.f;
    for (int k = 0; k < D_; ++k) s += xb[k] * w[k];
    meanq[o] = s;
}

// ------------------------------ gate / wr MLPs -----------------------------
// one block per (b,h); 256 threads
__global__ __launch_bounds__(256)
void mlp_kernel(const float* __restrict__ meanq,
                const float* __restrict__ g_ln_w, const float* __restrict__ g_ln_b,
                const float* __restrict__ g_w1,   const float* __restrict__ g_b1,
                const float* __restrict__ g_w2,   const float* __restrict__ g_b2,
                const float* __restrict__ w_ln_w, const float* __restrict__ w_ln_b,
                const float* __restrict__ w_w1,   const float* __restrict__ w_b1,
                const float* __restrict__ w_w2,   const float* __restrict__ w_b2,
                float* __restrict__ graw,   // [B*H, 2050]
                float* __restrict__ scale)  // [B*H, 64] = 1 + sigmoid(...)
{
    __shared__ float mq[DH_], hg[DH_], hw[DH_], h1[4 * DH_], t1[DH_];
    int bh = blockIdx.x, tid = threadIdx.x;
    const float* m = meanq + (size_t)bh * DH_;
    if (tid < DH_) mq[tid] = m[tid];
    __syncthreads();

    float mu = 0.f;
    for (int i = 0; i < DH_; ++i) mu += mq[i];
    mu *= (1.0f / DH_);
    float var = 0.f;
    for (int i = 0; i < DH_; ++i) { float d = mq[i] - mu; var += d * d; }
    var *= (1.0f / DH_);
    float rstd = rsqrtf(var + 1e-5f);

    if (tid < DH_) {
        float nrm = (mq[tid] - mu) * rstd;
        hg[tid] = nrm * g_ln_w[tid] + g_ln_b[tid];
        hw[tid] = nrm * w_ln_w[tid] + w_ln_b[tid];
    }
    __syncthreads();

    {   // gate hidden: 256 outputs
        float s = g_b1[tid];
        const float* w = g_w1 + (size_t)tid * DH_;
        for (int i = 0; i < DH_; ++i) s += hg[i] * w[i];
        h1[tid] = gelu_exact(s);
    }
    if (tid < DH_) {   // wr hidden: 64 outputs
        float s = w_b1[tid];
        const float* w = w_w1 + (size_t)tid * DH_;
        for (int i = 0; i < DH_; ++i) s += hw[i] * w[i];
        t1[tid] = gelu_exact(s);
    }
    __syncthreads();

    // gate output: 2050 values (real[0:1025] | imag[0:1025])
    float* go = graw + (size_t)bh * (2 * NF_);
    for (int o = tid; o < 2 * NF_; o += 256) {
        float s = g_b2[o];
        const float* w = g_w2 + (size_t)o * (4 * DH_);
        for (int i = 0; i < 4 * DH_; ++i) s += h1[i] * w[i];
        go[o] = s;
    }
    if (tid < DH_) {   // wr output -> (1 + sigmoid)
        float s = w_b2[tid];
        const float* w = w_w2 + (size_t)tid * DH_;
        for (int i = 0; i < DH_; ++i) s += t1[i] * w[i];
        scale[(size_t)bh * DH_ + tid] = 1.0f + 1.0f / (1.0f + __expf(-s));
    }
}

// gmean[b, f] = mean over h of graw[(b*16+h), f]     (8 * 2050 outputs)
__global__ void gmean_kernel(const float* __restrict__ graw,
                             float* __restrict__ gmean) {
    int i = blockIdx.x * blockDim.x + threadIdx.x;
    if (i >= B_ * 2 * NF_) return;
    int b = i / (2 * NF_), f = i % (2 * NF_);
    float s = 0.f;
    for (int h = 0; h < H_; ++h) s += graw[(size_t)(b * H_ + h) * (2 * NF_) + f];
    gmean[i] = s * (1.0f / H_);
}

// ------------------------------ WMMA bf16 GEMM -----------------------------
// C[M,N] = A[M,K] (bf16, row-major) * W[N,K]^T (bf16, row-major).
// Block = 256 threads = 8 waves; block tile 64x64; wave tile 16x32.
// TRANS_OUT=true: store C transposed per-batch into (B, N, L) f32 layout
// (rows are b*2048 + l), feeding the FFT with contiguous sequences.
template <bool TRANS_OUT>
__global__ __launch_bounds__(256)
void gemm_bf16_nt(const __bf16* __restrict__ A, const __bf16* __restrict__ W,
                  float* __restrict__ C, int M, int N, int K) {
    int ntn  = N >> 6;
    int m0   = (blockIdx.x / ntn) << 6;
    int n0   = (blockIdx.x % ntn) << 6;
    int wave = threadIdx.x >> 5;
    int lane = threadIdx.x & 31;
    int mrow = m0 + (wave & 3) * 16;
    int ncol = n0 + (wave >> 2) * 32;
    int lm   = lane & 15;
    int lk   = (lane >> 4) << 4;   // 0 or 16 within the K=32 slab

    const __bf16* Ap = A + (size_t)(mrow + lm) * K + lk;
    const __bf16* W0 = W + (size_t)(ncol + lm) * K + lk;
    const __bf16* W1 = W0 + (size_t)16 * K;

    v8f acc0 = {}; v8f acc1 = {};
    for (int k = 0; k < K; k += 32) {
        __builtin_prefetch(Ap + k + 256, 0, 1);
        __builtin_prefetch(W0 + k + 256, 0, 1);
        v16bf a  = *(const v16bf*)(Ap + k);
        v16bf b0 = *(const v16bf*)(W0 + k);
        v16bf b1 = *(const v16bf*)(W1 + k);
        acc0 = __builtin_amdgcn_wmma_f32_16x16x32_bf16(false, a, false, b0,
                                                       (short)0, acc0, false, false);
        acc1 = __builtin_amdgcn_wmma_f32_16x16x32_bf16(false, a, false, b1,
                                                       (short)0, acc1, false, false);
    }

    // C/D layout: VGPR r of lane -> M = r + 8*(lane>=16), N = lane&15
    if constexpr (TRANS_OUT) {
        int bidx  = mrow >> 11;                               // batch (rows = b*L + l)
        int lbase = (mrow & (L_ - 1)) + ((lane >> 4) << 3);
        float* p0 = C + ((size_t)(bidx * D_ + ncol + lm) * L_) + lbase;
        float* p1 = C + ((size_t)(bidx * D_ + ncol + 16 + lm) * L_) + lbase;
        #pragma unroll
        for (int r = 0; r < 8; ++r) { p0[r] = acc0[r]; p1[r] = acc1[r]; }
    } else {
        int rbase = mrow + ((lane >> 4) << 3);
        #pragma unroll
        for (int r = 0; r < 8; ++r) {
            C[(size_t)(rbase + r) * N + (ncol + lm)]      = acc0[r];
            C[(size_t)(rbase + r) * N + (ncol + 16 + lm)] = acc1[r];
        }
    }
}

// ------------------------------ FFT kernel ---------------------------------
// One block per sequence seq = b*1024 + (h*64+d); 256 threads; Stockham DIF
// complex FFT (N=2048) in LDS, pointwise * Hermitian-extended G, inverse FFT,
// writeback * (1/N) * (1 + sigmoid(wr gate)) as bf16 (still (B,D,L) layout).
__global__ __launch_bounds__(256)
void fft_kernel(const float* __restrict__ vt, const float* __restrict__ gmean,
                const float* __restrict__ scale, __bf16* __restrict__ yt) {
    __shared__ float re0[L_], im0[L_], re1[L_], im1[L_];
    const int seq = blockIdx.x;
    const int b   = seq >> 10;
    const int tid = threadIdx.x;
    const float PI = 3.14159265358979323846f;

    const float* src = vt + (size_t)seq * L_;
    for (int i = tid; i < L_; i += 256) { re0[i] = src[i]; im0[i] = 0.f; }
    __syncthreads();

    float *sr = re0, *si = im0, *dr = re1, *di = im1;

    // forward FFT
    int l = L_ / 2, m = 1;
    for (int s = 0; s < 11; ++s) {
        for (int t = tid; t < L_ / 2; t += 256) {
            int j = t / m, k = t - j * m;
            float ar = sr[j * m + k],        ai = si[j * m + k];
            float br = sr[(j + l) * m + k],  bi = si[(j + l) * m + k];
            float sn, c;
            __sincosf(-PI * (float)j / (float)l, &sn, &c);
            float xr = ar - br, xi = ai - bi;
            dr[(2 * j) * m + k]     = ar + br;
            di[(2 * j) * m + k]     = ai + bi;
            dr[(2 * j + 1) * m + k] = xr * c - xi * sn;
            di[(2 * j + 1) * m + k] = xr * sn + xi * c;
        }
        __syncthreads();
        float* t0 = sr; sr = dr; dr = t0;
        float* t1 = si; si = di; di = t1;
        l >>= 1; m <<= 1;
    }

    // pointwise multiply by G (Hermitian extension of g[b])
    const float* grp = gmean + (size_t)b * (2 * NF_);
    for (int k = tid; k < L_; k += 256) {
        float Gr, Gi;
        if (k <= L_ / 2) { Gr = grp[k];      Gi = grp[NF_ + k]; }
        else             { Gr = grp[L_ - k]; Gi = -grp[NF_ + (L_ - k)]; }
        float xr = sr[k], xi = si[k];
        sr[k] = xr * Gr - xi * Gi;
        si[k] = xr * Gi + xi * Gr;
    }
    __syncthreads();

    // inverse FFT (conjugate twiddles; 1/N folded into writeback)
    l = L_ / 2; m = 1;
    for (int s = 0; s < 11; ++s) {
        for (int t = tid; t < L_ / 2; t += 256) {
            int j = t / m, k = t - j * m;
            float ar = sr[j * m + k],        ai = si[j * m + k];
            float br = sr[(j + l) * m + k],  bi = si[(j + l) * m + k];
            float sn, c;
            __sincosf(PI * (float)j / (float)l, &sn, &c);
            float xr = ar - br, xi = ai - bi;
            dr[(2 * j) * m + k]     = ar + br;
            di[(2 * j) * m + k]     = ai + bi;
            dr[(2 * j + 1) * m + k] = xr * c - xi * sn;
            di[(2 * j + 1) * m + k] = xr * sn + xi * c;
        }
        __syncthreads();
        float* t0 = sr; sr = dr; dr = t0;
        float* t1 = si; si = di; di = t1;
        l >>= 1; m <<= 1;
    }

    // Haar+gate path collapses to y *= (1 + sigmoid(wr)):  recon == gate*y.
    const float g1 = scale[seq] * (1.0f / (float)L_);
    __bf16* dst = yt + (size_t)seq * L_;
    for (int i = tid; i < L_; i += 256) dst[i] = (__bf16)(sr[i] * g1);
}

// ------------------------------ tiled transpose ----------------------------
template <typename T>
__global__ void transpose_kernel(const T* __restrict__ in, T* __restrict__ out,
                                 int rows, int cols) {
    __shared__ T tile[32][33];
    size_t base = (size_t)blockIdx.z * rows * cols;
    int r0 = blockIdx.y * 32, c0 = blockIdx.x * 32;
    for (int i = threadIdx.y; i < 32; i += 8)
        tile[i][threadIdx.x] = in[base + (size_t)(r0 + i) * cols + (c0 + threadIdx.x)];
    __syncthreads();
    for (int i = threadIdx.y; i < 32; i += 8)
        out[base + (size_t)(c0 + i) * rows + (r0 + threadIdx.x)] = tile[threadIdx.x][i];
}

// ------------------------------ launcher -----------------------------------

extern "C" void kernel_launch(void* const* d_in, const int* in_sizes, int n_in,
                              void* d_out, int out_size, void* d_ws, size_t ws_size,
                              hipStream_t stream) {
    (void)in_sizes; (void)n_in; (void)out_size; (void)ws_size;

    const float* x       = (const float*)d_in[0];
    const float* Wq      = (const float*)d_in[1];
    const float* Wv      = (const float*)d_in[2];
    const float* Wo      = (const float*)d_in[3];
    const float* g_ln_w  = (const float*)d_in[4];
    const float* g_ln_b  = (const float*)d_in[5];
    const float* g_w1    = (const float*)d_in[6];
    const float* g_b1    = (const float*)d_in[7];
    const float* g_w2    = (const float*)d_in[8];
    const float* g_b2    = (const float*)d_in[9];
    const float* w_ln_w  = (const float*)d_in[10];
    const float* w_ln_b  = (const float*)d_in[11];
    const float* w_w1    = (const float*)d_in[12];
    const float* w_b1    = (const float*)d_in[13];
    const float* w_w2    = (const float*)d_in[14];
    const float* w_b2    = (const float*)d_in[15];
    float* out = (float*)d_out;

    // workspace bump allocator (256B aligned)
    char* base = (char*)d_ws;
    size_t off = 0;
    auto alloc = [&](size_t bytes) -> char* {
        char* p = base + off;
        off += (bytes + 255) & ~(size_t)255;
        return p;
    };
    __bf16* xbf   = (__bf16*)alloc((size_t)M_ * D_ * 2);   // 32 MB
    __bf16* wvbf  = (__bf16*)alloc((size_t)D_ * D_ * 2);   //  2 MB
    __bf16* wobf  = (__bf16*)alloc((size_t)D_ * D_ * 2);   //  2 MB
    float*  xbar  = (float*) alloc((size_t)B_ * D_ * 4);
    float*  meanq = (float*) alloc((size_t)B_ * D_ * 4);
    float*  graw  = (float*) alloc((size_t)B_ * H_ * 2 * NF_ * 4);
    float*  gmean = (float*) alloc((size_t)B_ * 2 * NF_ * 4);
    float*  scale = (float*) alloc((size_t)B_ * H_ * DH_ * 4);
    float*  vt    = (float*) alloc((size_t)B_ * D_ * L_ * 4);   // 64 MB (B,D,L)
    __bf16* yt    = (__bf16*)alloc((size_t)B_ * D_ * L_ * 2);   // 32 MB (B,D,L)
    __bf16* ybf   = (__bf16*)vt;   // reuse vt after FFT consumed it: (B,L,D) bf16

    // 1) casts to bf16
    cast_f32_to_bf16<<<4096, 256, 0, stream>>>(x,  xbf,  M_ * D_);
    cast_f32_to_bf16<<<1024, 256, 0, stream>>>(Wv, wvbf, D_ * D_);
    cast_f32_to_bf16<<<1024, 256, 0, stream>>>(Wo, wobf, D_ * D_);

    // 2) mean_q via xbar (q-GEMM eliminated analytically)
    colmean_kernel<<<(B_ * D_ + 255) / 256, 256, 0, stream>>>(x, xbar);
    meanq_kernel  <<<(B_ * D_ + 255) / 256, 256, 0, stream>>>(xbar, Wq, meanq);

    // 3) gate + write-gate MLPs, then mean over heads
    mlp_kernel<<<B_ * H_, 256, 0, stream>>>(meanq,
        g_ln_w, g_ln_b, g_w1, g_b1, g_w2, g_b2,
        w_ln_w, w_ln_b, w_w1, w_b1, w_w2, w_b2, graw, scale);
    gmean_kernel<<<(B_ * 2 * NF_ + 255) / 256, 256, 0, stream>>>(graw, gmean);

    // 4) v = x @ Wv^T  (WMMA bf16), stored transposed into (B, D, L) f32
    gemm_bf16_nt<true><<<(M_ / 64) * (D_ / 64), 256, 0, stream>>>(
        xbf, wvbf, vt, M_, D_, D_);

    // 5) FFT * G * IFFT * (1+gate)  -> yt (B,D,L) bf16
    fft_kernel<<<B_ * D_, 256, 0, stream>>>(vt, gmean, scale, yt);

    // 6) transpose yt (B,D,L) -> ybf (B,L,D) bf16 for the output GEMM
    {
        dim3 g(L_ / 32, D_ / 32, B_), t(32, 8);
        transpose_kernel<__bf16><<<g, t, 0, stream>>>(yt, ybf, D_, L_);
    }

    // 7) out = y @ Wo^T  (WMMA bf16, f32 result)
    gemm_bf16_nt<false><<<(M_ / 64) * (D_ / 64), 256, 0, stream>>>(
        ybf, wobf, out, M_, D_, D_);
}
